// Encoder_20426864460305
// MI455X (gfx1250) — matrix operand order repaired and checked by reading, not compiled, not run on hardware
//
#include <hip/hip_runtime.h>
#include <math.h>

// ---------------------------------------------------------------------------
// GRU: xp = x@W + b0 (parallel GEMM), then 512-step scan with rec = h@U + b1.
// Scan kernel is persistent: 32 WGs, each pins a 96-column slice of U
// (1024x96 bf16 = 192KB) in LDS; device-wide barrier per timestep.
// bf16 in / f32 accumulate via v_wmma_f32_16x16x32_bf16 (wave32).
// xp gate inputs for step t+1 are prefetched into LDS with
// global_load_async_to_lds_b128 (ASYNCcnt) while step t's GEMM runs.
// ---------------------------------------------------------------------------

typedef __attribute__((ext_vector_type(16))) __bf16        v16bf;
typedef __attribute__((ext_vector_type(8)))  float         v8f;
typedef __attribute__((ext_vector_type(4)))  unsigned int  v4u;

union BfPack { v4u u2[2]; v16bf v; };

static __device__ __forceinline__ unsigned short f2bf(float f) {
  unsigned u = __float_as_uint(f);
  u += 0x7FFFu + ((u >> 16) & 1u);            // round-to-nearest-even
  return (unsigned short)(u >> 16);
}
static __device__ __forceinline__ float bf2f(unsigned short h) {
  return __uint_as_float(((unsigned)h) << 16);
}

// Two 16-byte chunks -> one 16-element bf16 WMMA operand.
static __device__ __forceinline__ v16bf ld2(const unsigned short* p0,
                                            const unsigned short* p1) {
  BfPack pk;
  pk.u2[0] = *(const v4u*)p0;
  pk.u2[1] = *(const v4u*)p1;
  return pk.v;
}
// A operand (16x32, MxK): lane<16 holds K {0..7,16..23}, lane>=16 {8..15,24..31}
static __device__ __forceinline__ v16bf ld_a(const unsigned short* row, int kk, int half) {
  const unsigned short* p = row + kk + half * 8;
  return ld2(p, p + 16);
}
// B operand (32x16, KxN): lane<16 holds K {0..7,16..23}, lane>=16 {8..15,24..31}
static __device__ __forceinline__ v16bf ld_b(const unsigned short* col, int kk, int half) {
  const unsigned short* p = col + kk + half * 8;
  return ld2(p, p + 16);
}

static __device__ __forceinline__ void grid_barrier(unsigned* cnt, unsigned target) {
  __syncthreads();
  if (threadIdx.x == 0) {
    __threadfence();                 // publish h stores device-wide
    atomicAdd(cnt, 1u);
    volatile unsigned* vc = cnt;
    while (*vc < target) { __builtin_amdgcn_s_sleep(1); }
    __threadfence();
  }
  __syncthreads();
}

// Async copy: 16B from each lane's global address into each lane's LDS offset.
static __device__ __forceinline__ void async_b128(unsigned lds_off, const float* src) {
  asm volatile("global_load_async_to_lds_b128 %0, %1, off"
               :: "v"(lds_off), "v"(src) : "memory");
}

#define GDIM   3072
#define HID    1024
#define BATCH  64
#define SEQ    512
#define NWG2   32          // persistent workgroups in scan kernel

// ---------------------------------------------------------------------------
// Kernel 1: xp[t][b][g] = x[b][t][:] @ W[:, g] + b0[g]
// Block tile 64(M) x 128(N), K=256 staged in LDS as bf16, 8 waves/block.
// ---------------------------------------------------------------------------
__global__ __launch_bounds__(256) void xproj_kernel(
    const float* __restrict__ x, const float* __restrict__ W,
    const float* __restrict__ bias, float* __restrict__ xp)
{
  extern __shared__ unsigned short smem1[];
  unsigned short* As = smem1;              // [64][256]  (row-major, MxK)
  unsigned short* Bs = smem1 + 64 * 256;   // [128][256] (transposed: [n][k])

  const int tid   = threadIdx.x;
  const int nBase = blockIdx.x * 128;
  const int mBase = blockIdx.y * 64;

  // Stage A: x rows (coalesced along K)
  for (int i = 0; i < 64; ++i)
    As[i * 256 + tid] = f2bf(x[(size_t)(mBase + i) * 256 + tid]);
  // Stage B: W transposed into [n][k] (coalesced along g)
  for (int i = 0; i < 128; ++i) {
    int k = i * 2 + (tid >> 7);
    int c = tid & 127;
    Bs[c * 256 + k] = f2bf(W[(size_t)k * GDIM + nBase + c]);
  }
  __syncthreads();

  const int w    = tid >> 5;
  const int lane = tid & 31;
  const int half = lane >> 4;
  const int l16  = lane & 15;
  const int mT   = w & 3;      // 4 M-tiles
  const int nG   = w >> 2;     // 2 groups x 4 N-tiles = 8 N-tiles

  const unsigned short* aRow = As + (mT * 16 + l16) * 256;
  const unsigned short* bCol[4];
#pragma unroll
  for (int nt = 0; nt < 4; ++nt)
    bCol[nt] = Bs + ((nG * 4 + nt) * 16 + l16) * 256;

  v8f acc[4];
#pragma unroll
  for (int nt = 0; nt < 4; ++nt)
#pragma unroll
    for (int e = 0; e < 8; ++e) acc[nt][e] = 0.0f;

  // 1-stage software pipeline: issue k+32 loads before k's WMMAs.
  v16bf aC = ld_a(aRow, 0, half);
  v16bf bC0 = ld_b(bCol[0], 0, half), bC1 = ld_b(bCol[1], 0, half);
  v16bf bC2 = ld_b(bCol[2], 0, half), bC3 = ld_b(bCol[3], 0, half);
#pragma unroll
  for (int kk = 0; kk < 256; kk += 32) {
    v16bf aN = aC, bN0 = bC0, bN1 = bC1, bN2 = bC2, bN3 = bC3;
    if (kk + 32 < 256) {
      aN  = ld_a(aRow, kk + 32, half);
      bN0 = ld_b(bCol[0], kk + 32, half);
      bN1 = ld_b(bCol[1], kk + 32, half);
      bN2 = ld_b(bCol[2], kk + 32, half);
      bN3 = ld_b(bCol[3], kk + 32, half);
    }
    acc[0] = __builtin_amdgcn_wmma_f32_16x16x32_bf16(false, aC, false, bC0, (short)0, acc[0], false, false);
    acc[1] = __builtin_amdgcn_wmma_f32_16x16x32_bf16(false, aC, false, bC1, (short)0, acc[1], false, false);
    acc[2] = __builtin_amdgcn_wmma_f32_16x16x32_bf16(false, aC, false, bC2, (short)0, acc[2], false, false);
    acc[3] = __builtin_amdgcn_wmma_f32_16x16x32_bf16(false, aC, false, bC3, (short)0, acc[3], false, false);
    aC = aN; bC0 = bN0; bC1 = bN1; bC2 = bN2; bC3 = bN3;
  }

  // C layout: VGPR r -> M = r + 8*half, N = lane&15
#pragma unroll
  for (int nt = 0; nt < 4; ++nt) {
    int g  = nBase + (nG * 4 + nt) * 16 + l16;
    float bv = bias[g];                         // b[0][g]
#pragma unroll
    for (int r = 0; r < 8; ++r) {
      int rowM = mBase + mT * 16 + r + 8 * half; // rowM = b*512 + t
      int bb = rowM >> 9;
      int tt = rowM & 511;
      xp[((size_t)tt * BATCH + bb) * GDIM + g] = acc[nt][r] + bv;
    }
  }
}

// ---------------------------------------------------------------------------
// Kernel 2: persistent GRU scan. WG wg owns hidden units [wg*32, wg*32+32).
// LDS: U slice transposed [96][1024] bf16 (192KB) + rec [64][96] f32 (24KB)
//      + double-buffered xp gate slice [2][64][96] f32 (48KB).
// ---------------------------------------------------------------------------
__global__ __launch_bounds__(256) void gru_scan_kernel(
    const float* __restrict__ hidden0, const float* __restrict__ U,
    const float* __restrict__ bias,    const float* __restrict__ xp,
    unsigned short* __restrict__ hbuf0, unsigned short* __restrict__ hbuf1,
    unsigned* __restrict__ cnt, float* __restrict__ out)
{
  extern __shared__ unsigned short smem2[];
  unsigned short* Us  = smem2;                              // [96][1024] bf16
  float*          rec = (float*)(smem2 + 96 * 1024);        // [64][96]   f32
  float*          xpl = rec + BATCH * 96;                   // [2][64][96] f32

  const int tid = threadIdx.x;
  const int wg  = blockIdx.x;
  const int j0  = wg * 32;

  // Load U slice (f32 -> bf16), transposed to [c][k]
  for (int idx = tid; idx < 96 * 1024; idx += 256) {
    int k = idx / 96;
    int c = idx - k * 96;
    int gcol = (c >> 5) * HID + j0 + (c & 31);
    Us[c * 1024 + k] = f2bf(U[(size_t)k * GDIM + gcol]);
  }
  // Init own slice of h (buffer 0)
  for (int idx = tid; idx < BATCH * 32; idx += 256) {
    int b = idx >> 5;
    int j = j0 + (idx & 31);
    hbuf0[b * HID + j] = f2bf(hidden0[(size_t)b * HID + j]);
  }

  // Prefetch xp for t=0 into xpl buffer 0 (6 async b128 per thread = 24KB).
  const unsigned xplOff = (unsigned)(size_t)(void*)xpl;     // LDS byte offset
  {
    const int tp = 0;
#pragma unroll
    for (int i = 0; i < 6; ++i) {
      int q = tid + i * 256;                 // 0..1535 lane-chunks of 16B
      int b = q / 24;
      int r = q - b * 24;
      int gate = r >> 3;
      int quad = r & 7;
      const float* src = xp + ((size_t)tp * BATCH + b) * GDIM + gate * HID + j0 + quad * 4;
      unsigned dst = xplOff + (unsigned)(((0 * BATCH + b) * 96 + gate * 32 + quad * 4) * 4);
      async_b128(dst, src);
    }
  }

  grid_barrier(cnt, NWG2);   // all slices of h + all U tiles ready

  const int w    = tid >> 5;
  const int lane = tid & 31;
  const int half = lane >> 4;
  const int l16  = lane & 15;
  const int mT   = w & 3;    // 4 M-tiles over batch=64
  const int nG   = w >> 2;   // 2 groups x 3 N-tiles = 6 N-tiles over 96 cols

  const unsigned short* bCol[3];
#pragma unroll
  for (int nt = 0; nt < 3; ++nt)
    bCol[nt] = Us + ((nG * 3 + nt) * 16 + l16) * 1024;

  for (int t = 0; t < SEQ; ++t) {
    const unsigned short* hcur  = (t & 1) ? hbuf1 : hbuf0;
    unsigned short*       hnext = (t & 1) ? hbuf0 : hbuf1;

    // Issue async prefetch of xp for step t+1 (dummy t=0 refetch at the end
    // keeps exactly 6 ops in flight so the <=6 wait below stays correct).
    {
      const int tp  = (t + 1 < SEQ) ? (t + 1) : 0;
      const int buf = (t + 1) & 1;
#pragma unroll
      for (int i = 0; i < 6; ++i) {
        int q = tid + i * 256;
        int b = q / 24;
        int r = q - b * 24;
        int gate = r >> 3;
        int quad = r & 7;
        const float* src = xp + ((size_t)tp * BATCH + b) * GDIM + gate * HID + j0 + quad * 4;
        unsigned dst = xplOff + (unsigned)(((buf * BATCH + b) * 96 + gate * 32 + quad * 4) * 4);
        async_b128(dst, src);
      }
    }

    // rec = h @ U_slice, 1-stage software pipelined.
    v8f acc[3];
#pragma unroll
    for (int nt = 0; nt < 3; ++nt)
#pragma unroll
      for (int e = 0; e < 8; ++e) acc[nt][e] = 0.0f;

    const unsigned short* aRow = hcur + (mT * 16 + l16) * HID;
    v16bf aC  = ld_a(aRow, 0, half);
    v16bf bC0 = ld_b(bCol[0], 0, half);
    v16bf bC1 = ld_b(bCol[1], 0, half);
    v16bf bC2 = ld_b(bCol[2], 0, half);
#pragma unroll
    for (int kk = 0; kk < HID; kk += 32) {
      v16bf aN = aC, bN0 = bC0, bN1 = bC1, bN2 = bC2;
      if (kk + 32 < HID) {
        aN  = ld_a(aRow, kk + 32, half);       // global (L2-hot)
        bN0 = ld_b(bCol[0], kk + 32, half);    // ds_load_b128
        bN1 = ld_b(bCol[1], kk + 32, half);
        bN2 = ld_b(bCol[2], kk + 32, half);
      }
      acc[0] = __builtin_amdgcn_wmma_f32_16x16x32_bf16(false, aC, false, bC0, (short)0, acc[0], false, false);
      acc[1] = __builtin_amdgcn_wmma_f32_16x16x32_bf16(false, aC, false, bC1, (short)0, acc[1], false, false);
      acc[2] = __builtin_amdgcn_wmma_f32_16x16x32_bf16(false, aC, false, bC2, (short)0, acc[2], false, false);
      aC = aN; bC0 = bN0; bC1 = bN1; bC2 = bN2;
    }

#pragma unroll
    for (int nt = 0; nt < 3; ++nt) {
      int c = (nG * 3 + nt) * 16 + l16;
#pragma unroll
      for (int r = 0; r < 8; ++r) {
        int b = mT * 16 + r + 8 * half;
        rec[b * 96 + c] = acc[nt][r];
      }
    }

    // Step-t xp prefetch was issued one iteration ago; async loads complete
    // in order, so waiting for <=6 outstanding guarantees it has landed.
    asm volatile("s_wait_asynccnt 0x6" ::: "memory");
    __syncthreads();

    // Gate math for own hidden slice: 64 batch x 32 units = 2048 elems
    const float* xplCur = xpl + (t & 1) * (BATCH * 96);
    for (int p = tid; p < BATCH * 32; p += 256) {
      int b  = p >> 5;
      int jl = p & 31;
      int j  = j0 + jl;
      float xz = xplCur[b * 96 + jl];
      float xr = xplCur[b * 96 + 32 + jl];
      float xh = xplCur[b * 96 + 64 + jl];
      float rz = rec[b * 96 + jl]      + bias[GDIM + j];
      float rr = rec[b * 96 + 32 + jl] + bias[GDIM + HID + j];
      float rh = rec[b * 96 + 64 + jl] + bias[GDIM + 2 * HID + j];
      float ho = bf2f(hcur[b * HID + j]);
      float z  = 1.0f / (1.0f + __expf(-(xz + rz)));
      float r  = 1.0f / (1.0f + __expf(-(xr + rr)));
      float hh = tanhf(xh + r * rh);
      float hn = z * ho + (1.0f - z) * hh;
      out[((size_t)b * SEQ + t) * HID + j] = hn;
      hnext[b * HID + j] = f2bf(hn);
      if (t == SEQ - 1)
        out[(size_t)BATCH * SEQ * HID + (size_t)b * HID + j] = hn;
    }
    grid_barrier(cnt, (unsigned)NWG2 * (t + 2));
  }
}

// ---------------------------------------------------------------------------
extern "C" void kernel_launch(void* const* d_in, const int* in_sizes, int n_in,
                              void* d_out, int out_size, void* d_ws, size_t ws_size,
                              hipStream_t stream) {
  const float* x      = (const float*)d_in[0];  // (64,512,256)
  const float* hidden = (const float*)d_in[1];  // (64,1024)
  const float* W      = (const float*)d_in[2];  // (256,3072)
  const float* U      = (const float*)d_in[3];  // (1024,3072)
  const float* bias   = (const float*)d_in[4];  // (2,3072)
  float* out          = (float*)d_out;

  char* ws = (char*)d_ws;
  unsigned*       cnt   = (unsigned*)ws;                               // 256 B
  unsigned short* hbuf0 = (unsigned short*)(ws + 256);                 // 128 KB
  unsigned short* hbuf1 = (unsigned short*)(ws + 256 + BATCH*HID*2);   // 128 KB
  float*          xp    = (float*)(ws + 256 + 2*(size_t)BATCH*HID*2);  // 402 MB

  hipMemsetAsync(cnt, 0, 256, stream);

  // Kernel 1: input projection
  dim3 g1(GDIM / 128, (BATCH * SEQ) / 64);
  size_t lds1 = (size_t)(64 * 256 + 128 * 256) * sizeof(unsigned short); // 96 KB
  hipLaunchKernelGGL(xproj_kernel, g1, dim3(256), lds1, stream, x, W, bias, xp);

  // Kernel 2: persistent recurrent scan
  size_t lds2 = (size_t)96 * 1024 * 2      // U slice (bf16)
              + (size_t)BATCH * 96 * 4     // rec (f32)
              + (size_t)2 * BATCH * 96 * 4;// xp double buffer (f32) = 264 KB
  hipLaunchKernelGGL(gru_scan_kernel, dim3(NWG2), dim3(256), lds2, stream,
                     hidden, U, bias, xp, hbuf0, hbuf1, cnt, out);
}
